// EHNNTransformerConv_67619965108620
// MI455X (gfx1250) — compile-verified, matrix operand order checked by reading, not run on hardware
//
#include <hip/hip_runtime.h>
#include <math.h>

#define N_NODES 100000
#define N_EDGES 50000
#define CE (N_NODES + N_EDGES)
#define CE_PAD 150016                  // CE rounded up to multiple of 32 (wave M-tile)
#define STEP_E (N_NODES / 8 + 1)       // 12501
#define SCALE_L 0.17677669529663687f   // 1/sqrt(32)
#define B0 256                         // blocks for att0 reductions

typedef unsigned short u16;
typedef __attribute__((ext_vector_type(16))) __bf16 v16bf;
typedef __attribute__((ext_vector_type(8)))  float  v8f;
typedef __attribute__((ext_vector_type(4)))  unsigned int u32x4;

// ---------- helpers ----------
__device__ inline u16 f2bf_u(float x) {
    union { float f; unsigned u; } v; v.f = x;
    unsigned r = v.u + 0x7FFFu + ((v.u >> 16) & 1u);
    return (u16)(r >> 16);
}
__device__ inline float bf2f(u16 h) {
    union { unsigned u; float f; } v; v.u = ((unsigned)h) << 16;
    return v.f;
}
__device__ inline float sinpe(int pos, int d, int dim) {
    int p = d & ~1;
    float f = __expf(-(float)p * (9.210340371976184f / (float)dim)); // ln(10000)
    float a = (float)pos * f;
    return (d & 1) ? __cosf(a) : __sinf(a);
}
__device__ inline float wred_sum(float v) {
    #pragma unroll
    for (int o = 16; o > 0; o >>= 1) v += __shfl_xor(v, o, 32);
    return v;
}

union FragU { u32x4 u[2]; v16bf v; };

__device__ inline v8f wmma_bf16(v16bf a, v16bf b, v8f c) {
    return __builtin_amdgcn_wmma_f32_16x16x32_bf16(false, a, false, b, (short)0, c, false, false);
}

// ---------- weight convert + transpose: f32 [K,Nc] -> bf16 [Nc,K] ----------
__global__ void wconv(const float* __restrict__ W, u16* __restrict__ Wt, int K, int Nc) {
    int idx = blockIdx.x * blockDim.x + threadIdx.x;
    if (idx >= K * Nc) return;
    int k = idx / Nc, n = idx % Nc;
    Wt[(size_t)n * K + k] = f2bf_u(W[(size_t)k * Nc + n]);
}

// ---------- register-blocked WMMA bf16 GEMM ----------
// Each wave computes a 32x64 tile of C = act(A[M,K] @ Wt[Nc,K]^T + bias) (+resid).
// Per K-step(32): 2 A-frags + 4 B-frags (12 x b128 loads) feed 8 WMMAs.
template<bool RELU, bool RESID, bool OUTF, bool OUTB>
__global__ __launch_bounds__(256) void gemm_bf16(
    const u16* __restrict__ A, const u16* __restrict__ Wt,
    const float* __restrict__ bias, const float* resid,
    float* outF, u16* outB, int M, int K, int Nc)
{
    int lane = threadIdx.x & 31;
    int wg   = blockIdx.x * 8 + (threadIdx.x >> 5);
    int ntn  = Nc >> 6;                 // 64-wide N macro-tiles
    int tiles = (M >> 5) * ntn;         // 32-tall M macro-tiles
    if (wg >= tiles) return;
    int tm = wg / ntn, tn = wg % ntn;
    int r  = lane & 15;
    int kb = (lane >> 4) << 3;          // base K offset (0 or 8)

    const u16* pa0 = A  + (size_t)(tm * 32 + r) * K + kb;
    const u16* pa1 = pa0 + (size_t)16 * K;
    const u16* pb0 = Wt + (size_t)(tn * 64 + r) * K + kb;
    size_t bK = (size_t)16 * K;

    v8f acc[2][4];
    #pragma unroll
    for (int mi = 0; mi < 2; ++mi)
        #pragma unroll
        for (int ni = 0; ni < 4; ++ni) acc[mi][ni] = (v8f){};

    for (int k0 = 0; k0 < K; k0 += 32) {
        FragU a0, a1, b0, b1, b2, b3;
        a0.u[0] = *(const u32x4*)(pa0 + k0);      a0.u[1] = *(const u32x4*)(pa0 + k0 + 16);
        a1.u[0] = *(const u32x4*)(pa1 + k0);      a1.u[1] = *(const u32x4*)(pa1 + k0 + 16);
        const u16* pb = pb0 + k0;
        b0.u[0] = *(const u32x4*)(pb);            b0.u[1] = *(const u32x4*)(pb + 16);
        b1.u[0] = *(const u32x4*)(pb + bK);       b1.u[1] = *(const u32x4*)(pb + bK + 16);
        b2.u[0] = *(const u32x4*)(pb + 2 * bK);   b2.u[1] = *(const u32x4*)(pb + 2 * bK + 16);
        b3.u[0] = *(const u32x4*)(pb + 3 * bK);   b3.u[1] = *(const u32x4*)(pb + 3 * bK + 16);
        if (k0 + 32 < K) {
            __builtin_prefetch(pa0 + k0 + 32, 0, 3);
            __builtin_prefetch(pa1 + k0 + 32, 0, 3);
        }
        acc[0][0] = wmma_bf16(a0.v, b0.v, acc[0][0]);
        acc[0][1] = wmma_bf16(a0.v, b1.v, acc[0][1]);
        acc[0][2] = wmma_bf16(a0.v, b2.v, acc[0][2]);
        acc[0][3] = wmma_bf16(a0.v, b3.v, acc[0][3]);
        acc[1][0] = wmma_bf16(a1.v, b0.v, acc[1][0]);
        acc[1][1] = wmma_bf16(a1.v, b1.v, acc[1][1]);
        acc[1][2] = wmma_bf16(a1.v, b2.v, acc[1][2]);
        acc[1][3] = wmma_bf16(a1.v, b3.v, acc[1][3]);
    }

    int nlo = lane & 15;
    int rhi = (lane >> 4) << 3;
    #pragma unroll
    for (int mi = 0; mi < 2; ++mi) {
        int row0 = tm * 32 + mi * 16 + rhi;
        #pragma unroll
        for (int ni = 0; ni < 4; ++ni) {
            int n = tn * 64 + ni * 16 + nlo;
            float bv = bias[n];
            #pragma unroll
            for (int i = 0; i < 8; ++i) {
                int row = row0 + i;
                float v = acc[mi][ni][i] + bv;
                if (RELU)  v = fmaxf(v, 0.f);
                if (RESID) v += resid[(size_t)row * Nc + n];
                if (OUTF)  outF[(size_t)row * Nc + n] = v;
                if (OUTB)  outB[(size_t)row * Nc + n] = f2bf_u(v);
            }
        }
    }
}

// ---------- hypernetwork q + folded KQ ----------
__global__ __launch_bounds__(512) void qhyper(
    const float* __restrict__ qW0, const float* __restrict__ qb0,
    const float* __restrict__ qW1, const float* __restrict__ qb1,
    const float* __restrict__ qW2, const float* __restrict__ qb2,
    const float* __restrict__ kW,  const float* __restrict__ kb,
    float* __restrict__ KQ, float* __restrict__ c16)
{
    __shared__ float pe[2][64];
    __shared__ float h0[2][256];
    __shared__ float h1[2][256];
    __shared__ float qs[512];
    int t = threadIdx.x;
    if (t < 128) { int rr = t >> 6, d = t & 63; pe[rr][d] = sinpe(rr, d, 64); }
    __syncthreads();
    int rr = t >> 8, j = t & 255;
    float s = qb0[j];
    for (int k = 0; k < 64; ++k) s += pe[rr][k] * qW0[k * 256 + j];
    h0[rr][j] = s;
    __syncthreads();
    s = qb1[j];
    for (int k = 0; k < 256; ++k) s += h0[rr][k] * qW1[k * 256 + j];
    h1[rr][j] = fmaxf(s, 0.f);
    __syncthreads();
    s = qb2[j];
    for (int k = 0; k < 256; ++k) s += h1[rr][k] * qW2[k * 256 + j];
    qs[t] = s;
    __syncthreads();
    for (int idx = t; idx < 256 * 16; idx += 512) {
        int k = idx >> 4, o = idx & 15;
        int m = o >> 3, h = o & 7;
        const float* w  = kW + (size_t)k * 512 + m * 256 + h * 32;
        const float* qq = qs + m * 256 + h * 32;
        float a = 0.f;
        for (int d = 0; d < 32; ++d) a += w[d] * qq[d];
        KQ[k * 16 + o] = a;
    }
    if (t < 16) {
        int m = t >> 3, h = t & 7;
        const float* bb = kb + m * 256 + h * 32;
        const float* qq = qs + m * 256 + h * 32;
        float a = 0.f;
        for (int d = 0; d < 32; ++d) a += bb[d] * qq[d];
        c16[t] = a;
    }
}

// ---------- prep: outer LN + inner LN + concat pe1 (edges first, nodes after, zero pad) ----------
__global__ __launch_bounds__(256) void prep1(
    const float* __restrict__ x_v, const float* __restrict__ x_e,
    const int* __restrict__ edge_orders,
    const float* __restrict__ ln_g, const float* __restrict__ ln_b,
    const float* __restrict__ n1_g, const float* __restrict__ n1_b,
    float* __restrict__ X0, u16* __restrict__ A1)
{
    int row = blockIdx.x * 8 + (threadIdx.x >> 5);
    if (row >= CE_PAD) return;
    int lane = threadIdx.x & 31;
    if (row >= CE) {                       // zero-fill pad rows (deterministic tail)
        #pragma unroll
        for (int i = 0; i < 8; ++i) {
            int col = lane + 32 * i;
            X0[(size_t)row * 256 + col] = 0.f;
            A1[(size_t)row * 512 + col] = 0;
            A1[(size_t)row * 512 + 256 + col] = 0;
        }
        return;
    }
    const float* src; int order;
    if (row < N_EDGES) { src = x_e + (size_t)row * 256; order = edge_orders[row]; }
    else               { src = x_v + (size_t)(row - N_EDGES) * 256; order = 1; }
    float x[8], y[8];
    #pragma unroll
    for (int i = 0; i < 8; ++i) x[i] = src[lane + 32 * i];
    float s = 0.f, s2 = 0.f;
    #pragma unroll
    for (int i = 0; i < 8; ++i) { s += x[i]; s2 += x[i] * x[i]; }
    s = wred_sum(s) * (1.f / 256.f); s2 = wred_sum(s2) * (1.f / 256.f);
    float rs = rsqrtf(fmaxf(s2 - s * s, 0.f) + 1e-5f);
    #pragma unroll
    for (int i = 0; i < 8; ++i) {
        int col = lane + 32 * i;
        y[i] = (x[i] - s) * rs * ln_g[col] + ln_b[col];
        X0[(size_t)row * 256 + col] = y[i];
    }
    float t = 0.f, t2 = 0.f;
    #pragma unroll
    for (int i = 0; i < 8; ++i) { t += y[i]; t2 += y[i] * y[i]; }
    t = wred_sum(t) * (1.f / 256.f); t2 = wred_sum(t2) * (1.f / 256.f);
    float rt = rsqrtf(fmaxf(t2 - t * t, 0.f) + 1e-5f);
    #pragma unroll
    for (int i = 0; i < 8; ++i) {
        int col = lane + 32 * i;
        float tv = (y[i] - t) * rt * n1_g[col] + n1_b[col];
        A1[(size_t)row * 512 + col]       = f2bf_u(tv);
        A1[(size_t)row * 512 + 256 + col] = f2bf_u(sinpe(order, col, 256));
    }
}

// ---------- logits: Xb @ KQ * scale ----------
__global__ void logits_k(const u16* __restrict__ Xb, const float* __restrict__ KQ,
                         const float* __restrict__ c16, float* __restrict__ logits)
{
    int idx = blockIdx.x * blockDim.x + threadIdx.x;
    if (idx >= CE * 16) return;
    int c = idx >> 4, o = idx & 15;
    const u16* x = Xb + (size_t)c * 256;
    float a = c16[o];
    for (int k = 0; k < 256; ++k) a += bf2f(x[k]) * KQ[k * 16 + o];
    logits[idx] = a * SCALE_L;
}

// ---------- att0: deterministic tree-reduced global softmax ----------
__global__ __launch_bounds__(256) void att0_p1(const float* __restrict__ logits, float* p1max) {
    int t = threadIdx.x, h = t & 7;
    float m = -1e30f;
    for (int c = blockIdx.x * 32 + (t >> 3); c < CE; c += B0 * 32)
        m = fmaxf(m, logits[(size_t)c * 16 + h]);
    __shared__ float lds[256];
    lds[t] = m; __syncthreads();
    for (int s = 128; s >= 8; s >>= 1) { if (t < s) lds[t] = fmaxf(lds[t], lds[t + s]); __syncthreads(); }
    if (t < 8) p1max[blockIdx.x * 8 + t] = lds[t];
}
__global__ void att0_p2(const float* __restrict__ p1max, float* gmax) {
    int t = threadIdx.x;
    if (t >= 8) return;
    float m = -1e30f;
    for (int b = 0; b < B0; ++b) m = fmaxf(m, p1max[b * 8 + t]);
    gmax[t] = m;
}
__global__ __launch_bounds__(256) void att0_p3(const float* __restrict__ logits,
                                               const float* __restrict__ v_all,
                                               const float* __restrict__ gmax,
                                               float* pnum, float* pden) {
    int t = threadIdx.x, h = t >> 5, d = t & 31;
    float gm = gmax[h], acc = 0.f, ps = 0.f;
    for (int c = blockIdx.x; c < CE; c += B0) {
        float p = __expf(logits[(size_t)c * 16 + h] - gm);
        acc += p * v_all[(size_t)c * 256 + t];
        ps  += p;
    }
    pnum[blockIdx.x * 256 + t] = acc;
    if (d == 0) pden[blockIdx.x * 8 + h] = ps;
}
__global__ void att0_p4(const float* __restrict__ pnum, const float* __restrict__ pden,
                        float* att0_raw) {
    int t = threadIdx.x, h = t >> 5;
    float num = 0.f, den = 0.f;
    for (int b = 0; b < B0; ++b) { num += pnum[b * 256 + t]; den += pden[b * 8 + h]; }
    att0_raw[t] = num / den;
}

// ---------- att0 mode-MLP (single row, one block) ----------
__global__ __launch_bounds__(256) void att0_mlp(
    const float* __restrict__ att0_raw,
    const float* __restrict__ n2_g, const float* __restrict__ n2_b,
    const float* __restrict__ m2W1, const float* __restrict__ m2b1,
    const float* __restrict__ m2W2, const float* __restrict__ m2b2,
    float* __restrict__ att0f)
{
    __shared__ float a0[512];
    __shared__ float hh[256];
    __shared__ float red[16];
    int t = threadIdx.x;
    float x = att0_raw[t];
    float s = wred_sum(x), s2 = wred_sum(x * x);
    if ((t & 31) == 0) { red[t >> 5] = s; red[8 + (t >> 5)] = s2; }
    __syncthreads();
    float mu = 0.f, m2s = 0.f;
    for (int i = 0; i < 8; ++i) { mu += red[i]; m2s += red[8 + i]; }
    mu *= (1.f / 256.f); m2s *= (1.f / 256.f);
    float rs = rsqrtf(fmaxf(m2s - mu * mu, 0.f) + 1e-5f);
    a0[t]       = (x - mu) * rs * n2_g[t] + n2_b[t];
    a0[256 + t] = sinpe(0, t, 256);           // pe2[0] for att0 mode
    __syncthreads();
    float acc = m2b1[t];
    for (int k = 0; k < 512; ++k) acc += a0[k] * m2W1[k * 256 + t];
    hh[t] = fmaxf(acc, 0.f);
    __syncthreads();
    acc = m2b2[t];
    for (int k = 0; k < 256; ++k) acc += hh[k] * m2W2[k * 256 + t];
    att0f[t] = x + acc;
}

// ---------- att1: analytic sparse incidence softmax (<=8 edges + self per node) ----------
__global__ __launch_bounds__(256) void att1_k(const float* __restrict__ logits,
                                              const float* __restrict__ v_all,
                                              float* __restrict__ att1)
{
    size_t gid = (size_t)blockIdx.x * 256 + threadIdx.x;
    if (gid >= (size_t)N_NODES * 256) return;
    int rrow = (int)(gid >> 8);
    int col  = (int)(gid & 255);
    int h    = col >> 5;
    int cs[9]; float ls[9]; int cnt = 0;
    cs[cnt++] = N_EDGES + rrow;                         // self entry
    #pragma unroll
    for (int i = 0; i < 8; ++i) {
        int jj = rrow - i * STEP_E;
        jj %= N_NODES; if (jj < 0) jj += N_NODES;
        if (jj < N_EDGES) cs[cnt++] = jj;
    }
    float m = -1e30f;
    for (int i = 0; i < cnt; ++i) {
        ls[i] = logits[(size_t)cs[i] * 16 + 8 + h];
        m = fmaxf(m, ls[i]);
    }
    float den = 0.f;
    for (int i = 0; i < cnt; ++i) { ls[i] = __expf(ls[i] - m); den += ls[i]; }
    float acc = 0.f;
    for (int i = 0; i < cnt; ++i) acc += ls[i] * v_all[(size_t)cs[i] * 256 + col];
    att1[gid] = acc / den;
}

// ---------- LN + concat pe2[1] over att1 rows ----------
__global__ __launch_bounds__(256) void prep2(const float* __restrict__ att1,
                                             const float* __restrict__ g, const float* __restrict__ b,
                                             u16* __restrict__ A2)
{
    int row = blockIdx.x * 8 + (threadIdx.x >> 5);
    if (row >= N_NODES) return;
    int lane = threadIdx.x & 31;
    float x[8];
    #pragma unroll
    for (int i = 0; i < 8; ++i) x[i] = att1[(size_t)row * 256 + lane + 32 * i];
    float s = 0.f, s2 = 0.f;
    #pragma unroll
    for (int i = 0; i < 8; ++i) { s += x[i]; s2 += x[i] * x[i]; }
    s = wred_sum(s) * (1.f / 256.f); s2 = wred_sum(s2) * (1.f / 256.f);
    float rs = rsqrtf(fmaxf(s2 - s * s, 0.f) + 1e-5f);
    #pragma unroll
    for (int i = 0; i < 8; ++i) {
        int col = lane + 32 * i;
        A2[(size_t)row * 512 + col]       = f2bf_u((x[i] - s) * rs * g[col] + b[col]);
        A2[(size_t)row * 512 + 256 + col] = f2bf_u(sinpe(1, col, 256));
    }
}

// ---------- h = att1' + att0 (in place); t3 = LN(h,n3) ----------
__global__ __launch_bounds__(256) void h_comb(float* h, const float* __restrict__ att0f,
                                              const float* __restrict__ g, const float* __restrict__ b,
                                              u16* __restrict__ t3)
{
    int row = blockIdx.x * 8 + (threadIdx.x >> 5);
    if (row >= N_NODES) return;
    int lane = threadIdx.x & 31;
    float x[8];
    #pragma unroll
    for (int i = 0; i < 8; ++i) {
        int col = lane + 32 * i;
        x[i] = h[(size_t)row * 256 + col] + att0f[col];
        h[(size_t)row * 256 + col] = x[i];
    }
    float s = 0.f, s2 = 0.f;
    #pragma unroll
    for (int i = 0; i < 8; ++i) { s += x[i]; s2 += x[i] * x[i]; }
    s = wred_sum(s) * (1.f / 256.f); s2 = wred_sum(s2) * (1.f / 256.f);
    float rs = rsqrtf(fmaxf(s2 - s * s, 0.f) + 1e-5f);
    #pragma unroll
    for (int i = 0; i < 8; ++i) {
        int col = lane + 32 * i;
        t3[(size_t)row * 256 + col] = f2bf_u((x[i] - s) * rs * g[col] + b[col]);
    }
}

// ---------- out1 = x_v + h + bias_b; fbuf = LN(out1, fn) ----------
__global__ __launch_bounds__(256) void out1_k(const float* __restrict__ x_v,
                                              const float* __restrict__ h,
                                              const float* __restrict__ bias_b,
                                              const float* __restrict__ g, const float* __restrict__ b,
                                              float* __restrict__ out1, u16* __restrict__ fb)
{
    int row = blockIdx.x * 8 + (threadIdx.x >> 5);
    if (row >= N_NODES) return;
    int lane = threadIdx.x & 31;
    float x[8];
    #pragma unroll
    for (int i = 0; i < 8; ++i) {
        int col = lane + 32 * i;
        x[i] = x_v[(size_t)row * 256 + col] + h[(size_t)row * 256 + col] + bias_b[col];
        out1[(size_t)row * 256 + col] = x[i];
    }
    float s = 0.f, s2 = 0.f;
    #pragma unroll
    for (int i = 0; i < 8; ++i) { s += x[i]; s2 += x[i] * x[i]; }
    s = wred_sum(s) * (1.f / 256.f); s2 = wred_sum(s2) * (1.f / 256.f);
    float rs = rsqrtf(fmaxf(s2 - s * s, 0.f) + 1e-5f);
    #pragma unroll
    for (int i = 0; i < 8; ++i) {
        int col = lane + 32 * i;
        fb[(size_t)row * 256 + col] = f2bf_u((x[i] - s) * rs * g[col] + b[col]);
    }
}

// =====================================================================
extern "C" void kernel_launch(void* const* d_in, const int* in_sizes, int n_in,
                              void* d_out, int out_size, void* d_ws, size_t ws_size,
                              hipStream_t stream)
{
    const float* x_v   = (const float*)d_in[0];
    const float* x_e   = (const float*)d_in[1];
    const float* ln_g  = (const float*)d_in[2];
    const float* ln_b  = (const float*)d_in[3];
    const float* n1_g  = (const float*)d_in[4];
    const float* n1_b  = (const float*)d_in[5];
    const float* n2_g  = (const float*)d_in[6];
    const float* n2_b  = (const float*)d_in[7];
    const float* n3_g  = (const float*)d_in[8];
    const float* n3_b  = (const float*)d_in[9];
    const float* fn_g  = (const float*)d_in[10];
    const float* fn_b  = (const float*)d_in[11];
    const float* qW0   = (const float*)d_in[12];
    const float* qb0   = (const float*)d_in[13];
    const float* qW1   = (const float*)d_in[14];
    const float* qb1   = (const float*)d_in[15];
    const float* qW2   = (const float*)d_in[16];
    const float* qb2   = (const float*)d_in[17];
    const float* kW    = (const float*)d_in[18];
    const float* kb    = (const float*)d_in[19];
    const float* vW    = (const float*)d_in[20];
    const float* vb    = (const float*)d_in[21];
    const float* m1W1  = (const float*)d_in[22];
    const float* m1b1  = (const float*)d_in[23];
    const float* m1W2  = (const float*)d_in[24];
    const float* m1b2  = (const float*)d_in[25];
    const float* m2W1  = (const float*)d_in[26];
    const float* m2b1  = (const float*)d_in[27];
    const float* m2W2  = (const float*)d_in[28];
    const float* m2b2  = (const float*)d_in[29];
    const float* m3W1  = (const float*)d_in[30];
    const float* m3b1  = (const float*)d_in[31];
    const float* m3W2  = (const float*)d_in[32];
    const float* m3b2  = (const float*)d_in[33];
    const float* biasb = (const float*)d_in[34];
    const float* fW1   = (const float*)d_in[35];
    const float* fb1   = (const float*)d_in[36];
    const float* fW2   = (const float*)d_in[37];
    const float* fb2   = (const float*)d_in[38];
    const int* edge_orders = (const int*)d_in[39];
    (void)in_sizes; (void)n_in; (void)out_size; (void)ws_size;

    char* ws = (char*)d_ws;
    size_t off = 0;
    auto alloc = [&](size_t bytes) { size_t o = off; off += (bytes + 255) & ~(size_t)255; return o; };
    // big regions (lifetime-reused)
    size_t R1 = alloc((size_t)CE_PAD * 512 * 2);   // A1 -> v_all -> ffnmid (spans into R2)
    size_t R2 = alloc((size_t)CE_PAD * 256 * 4);   // X0 -> att1_raw/att1'/h
    size_t R3 = alloc((size_t)CE_PAD * 256 * 2);   // mid1 -> A2(start) -> mid3 -> out1(start)
    size_t R4 = alloc((size_t)CE_PAD * 256 * 2);   // Xb  -> A2/out1 tail
    size_t R5 = alloc((size_t)CE * 16 * 4);        // logits [CE,16]
    size_t R6 = alloc((size_t)N_NODES * 256 * 2);  // mid2 -> t3 -> fbuf
    // bf16 transposed weights
    size_t Wm1a = alloc(512 * 256 * 2), Wm1b = alloc(256 * 256 * 2), Wv = alloc(256 * 256 * 2);
    size_t Wm2a = alloc(512 * 256 * 2), Wm2b = alloc(256 * 256 * 2);
    size_t Wm3a = alloc(256 * 256 * 2), Wm3b = alloc(256 * 256 * 2);
    size_t Wf1  = alloc(256 * 1024 * 2), Wf2 = alloc(1024 * 256 * 2);
    // small
    size_t KQO = alloc(256 * 16 * 4), C16 = alloc(64);
    size_t P1M = alloc(B0 * 8 * 4), GMX = alloc(32);
    size_t PNM = alloc(B0 * 256 * 4), PDN = alloc(B0 * 8 * 4);
    size_t A0R = alloc(256 * 4), A0F = alloc(256 * 4);

    float* X0      = (float*)(ws + R2);
    u16*   A1      = (u16*)  (ws + R1);
    u16*   mid1    = (u16*)  (ws + R3);
    u16*   Xb      = (u16*)  (ws + R4);
    float* v_all   = (float*)(ws + R1);
    float* logits  = (float*)(ws + R5);
    float* att1p   = (float*)(ws + R2);
    u16*   A2      = (u16*)  (ws + R3);
    u16*   mid2    = (u16*)  (ws + R6);
    u16*   t3      = (u16*)  (ws + R6);
    u16*   mid3    = (u16*)  (ws + R3);
    float* out1    = (float*)(ws + R3);
    u16*   fbuf    = (u16*)  (ws + R6);
    u16*   ffnmid  = (u16*)  (ws + R1);
    float* outF    = (float*)d_out;

    // 1) weight conversion/transpose
    auto wc = [&](const float* W, size_t o, int K, int Nc) {
        wconv<<<(K * Nc + 255) / 256, 256, 0, stream>>>(W, (u16*)(ws + o), K, Nc);
    };
    wc(m1W1, Wm1a, 512, 256); wc(m1W2, Wm1b, 256, 256); wc(vW, Wv, 256, 256);
    wc(m2W1, Wm2a, 512, 256); wc(m2W2, Wm2b, 256, 256);
    wc(m3W1, Wm3a, 256, 256); wc(m3W2, Wm3b, 256, 256);
    wc(fW1, Wf1, 256, 1024);  wc(fW2, Wf2, 1024, 256);

    // 2) hypernetwork q + folded KQ
    qhyper<<<1, 512, 0, stream>>>(qW0, qb0, qW1, qb1, qW2, qb2, kW, kb,
                                  (float*)(ws + KQO), (float*)(ws + C16));

    // 3) prep (LN + LN + concat pe1), zero-fills pad rows
    prep1<<<(CE_PAD + 7) / 8, 256, 0, stream>>>(x_v, x_e, edge_orders, ln_g, ln_b, n1_g, n1_b, X0, A1);

    // 4) m1 MLP (residual X0) -> Xb (bf16) ; 5) v projection
    {
        int tiles = (CE_PAD / 32) * (256 / 64);
        gemm_bf16<true,  false, false, true><<<(tiles + 7) / 8, 256, 0, stream>>>(
            A1, (u16*)(ws + Wm1a), m1b1, nullptr, nullptr, mid1, CE_PAD, 512, 256);
        gemm_bf16<false, true,  false, true><<<(tiles + 7) / 8, 256, 0, stream>>>(
            mid1, (u16*)(ws + Wm1b), m1b2, X0, nullptr, Xb, CE_PAD, 256, 256);
        gemm_bf16<false, false, true,  false><<<(tiles + 7) / 8, 256, 0, stream>>>(
            Xb, (u16*)(ws + Wv), vb, nullptr, v_all, nullptr, CE_PAD, 256, 256);
    }

    // 6) logits = scale * (Xb @ KQ + c)
    logits_k<<<(CE * 16 + 255) / 256, 256, 0, stream>>>(Xb, (float*)(ws + KQO),
                                                        (float*)(ws + C16), logits);

    // 7) att0 global softmax (tree-reduced) + its mode-MLP
    att0_p1<<<B0, 256, 0, stream>>>(logits, (float*)(ws + P1M));
    att0_p2<<<1, 32, 0, stream>>>((float*)(ws + P1M), (float*)(ws + GMX));
    att0_p3<<<B0, 256, 0, stream>>>(logits, v_all, (float*)(ws + GMX),
                                    (float*)(ws + PNM), (float*)(ws + PDN));
    att0_p4<<<1, 256, 0, stream>>>((float*)(ws + PNM), (float*)(ws + PDN), (float*)(ws + A0R));
    att0_mlp<<<1, 256, 0, stream>>>((float*)(ws + A0R), n2_g, n2_b, m2W1, m2b1, m2W2, m2b2,
                                    (float*)(ws + A0F));

    // 8) att1 sparse incidence softmax
    att1_k<<<N_NODES, 256, 0, stream>>>(logits, v_all, att1p);

    // 9) m2 MLP on att1 (residual in place)
    prep2<<<(N_NODES + 7) / 8, 256, 0, stream>>>(att1p, n2_g, n2_b, A2);
    {
        int tiles = (N_NODES / 32) * (256 / 64);
        gemm_bf16<true,  false, false, true><<<(tiles + 7) / 8, 256, 0, stream>>>(
            A2, (u16*)(ws + Wm2a), m2b1, nullptr, nullptr, mid2, N_NODES, 512, 256);
        gemm_bf16<false, true,  true,  false><<<(tiles + 7) / 8, 256, 0, stream>>>(
            mid2, (u16*)(ws + Wm2b), m2b2, att1p, att1p, nullptr, N_NODES, 256, 256);
    }

    // 10) h = att1' + att0 ; m3 MLP residual in place
    h_comb<<<(N_NODES + 7) / 8, 256, 0, stream>>>(att1p, (float*)(ws + A0F), n3_g, n3_b, t3);
    {
        int tiles = (N_NODES / 32) * (256 / 64);
        gemm_bf16<true,  false, false, true><<<(tiles + 7) / 8, 256, 0, stream>>>(
            t3, (u16*)(ws + Wm3a), m3b1, nullptr, nullptr, mid3, N_NODES, 256, 256);
        gemm_bf16<false, true,  true,  false><<<(tiles + 7) / 8, 256, 0, stream>>>(
            mid3, (u16*)(ws + Wm3b), m3b2, att1p, att1p, nullptr, N_NODES, 256, 256);
    }

    // 11) out1 = x_v + h + bias_b ; FFN with residual -> d_out
    out1_k<<<(N_NODES + 7) / 8, 256, 0, stream>>>(x_v, att1p, biasb, fn_g, fn_b, out1, fbuf);
    {
        int tiles1 = (N_NODES / 32) * (1024 / 64);
        int tiles2 = (N_NODES / 32) * (256 / 64);
        gemm_bf16<true,  false, false, true><<<(tiles1 + 7) / 8, 256, 0, stream>>>(
            fbuf, (u16*)(ws + Wf1), fb1, nullptr, nullptr, ffnmid, N_NODES, 256, 1024);
        gemm_bf16<false, true,  true,  false><<<(tiles2 + 7) / 8, 256, 0, stream>>>(
            ffnmid, (u16*)(ws + Wf2), fb2, out1, outF, nullptr, N_NODES, 1024, 256);
    }
}
